// AttentionLayer_38482906972715
// MI455X (gfx1250) — matrix operand order repaired and checked by reading, not compiled, not run on hardware
//
#include <hip/hip_runtime.h>
#include <hip/hip_bf16.h>
#include <math.h>

typedef __attribute__((ext_vector_type(16))) _Float16 v16h;
typedef _Float16 v16h_a16 __attribute__((ext_vector_type(16), aligned(16)));
typedef __attribute__((ext_vector_type(8)))  float    v8f;

constexpr int BB = 32;
constexpr int SS = 4096;
constexpr int DD = 256;

// Native CDNA5 transcendental tanh. v_nop after the TRANS op covers the
// "1 independent op before output use" hazard the compiler can't see.
__device__ __forceinline__ float tanh_hw(float x) {
    float y;
    asm("v_tanh_f32 %0, %1\n\tv_nop" : "=v"(y) : "v"(x));
    return y;
}

// ---------------------------------------------------------------------------
// Kernel 1: W [D][D] f32  ->  Wt [e][d] f16 (transposed, WMMA-B friendly)
// ---------------------------------------------------------------------------
__global__ __launch_bounds__(256) void convw_kernel(const float* __restrict__ W,
                                                    _Float16* __restrict__ Wt) {
    const int e = blockIdx.x;   // output column of original W
    const int d = threadIdx.x;  // K index
    Wt[e * DD + d] = (_Float16)W[d * DD + e];
}

// ---------------------------------------------------------------------------
// Kernel 2: fused  logits[row] = sum_n tanh( (X W)[row][n] + b[n] ) * V[n]
// 128 threads = 4 waves; each wave computes a 16-row tile with f16 WMMA.
// X is staged in LDS in natural (contiguous) k order so each
// lane's 16-half fragment slice is one contiguous 32B LDS read.
// ---------------------------------------------------------------------------
__global__ __launch_bounds__(128) void gemm_logits_kernel(
    const float*    __restrict__ X,      // [B*S, 256] f32
    const _Float16* __restrict__ Wt,     // [256(e)][256(d)] f16
    const float*    __restrict__ bias,   // [256]
    const float*    __restrict__ V,      // [256]
    float*          __restrict__ logits) // [B*S]
{
    constexpr int XS = 264;                        // padded half-stride per row
    __shared__ __align__(16) _Float16 xs[64 * XS]; // 64 rows, ~33.8 KB

    const int  tid  = threadIdx.x;
    const long row0 = (long)blockIdx.x * 64;

    // Cooperative load + f32->f16 convert of 64 rows (coalesced float4 reads),
    // storing in natural k order: halfs [kt*32 + khalf*16 .. +15] hold
    // k = khalf*16 + s, matching the contiguous B fragment.
    for (int idx = tid; idx < 64 * 64; idx += 128) {
        const int r  = idx >> 6;
        const int k  = (idx & 63) << 2;            // source k (4-half chunk)
        const float4 f = *(const float4*)(X + (row0 + r) * DD + k);
        const int tile = k >> 5;
        const int grp  = (k >> 3) & 3;             // 8-half group within tile
        const int perm = grp;
        _Float16* p = &xs[r * XS + tile * 32 + perm * 8 + (k & 7)];
        p[0] = (_Float16)f.x; p[1] = (_Float16)f.y;
        p[2] = (_Float16)f.z; p[3] = (_Float16)f.w;
    }
    __syncthreads();

    const int wave  = tid >> 5;
    const int lane  = tid & 31;
    const int mrow  = lane & 15;   // A: row M / B: col N / C: col N
    const int khalf = lane >> 4;   // half-wave selector per ISA layouts

    // A fragments: all 8 k-tiles resident in registers (one 32B read each).
    v16h a[8];
    const _Float16* xrow = &xs[(wave * 16 + mrow) * XS + khalf * 16];
    #pragma unroll
    for (int kt = 0; kt < 8; ++kt)
        a[kt] = *(const v16h_a16*)(xrow + kt * 32);

    float part[8];
    #pragma unroll
    for (int r = 0; r < 8; ++r) part[r] = 0.f;

    for (int nt = 0; nt < 16; ++nt) {
        v8f acc = {};
        // B fragment: lane col N = mrow, elems e -> K = khalf*16 + e (contiguous 32B)
        const _Float16* wb = Wt + ((nt * 16 + mrow) * DD + khalf * 16);
        #pragma unroll
        for (int kt = 0; kt < 8; ++kt) {
            const v16h bfrag = *(const v16h*)(wb + kt * 32);
            acc = __builtin_amdgcn_wmma_f32_16x16x32_f16(
                false, a[kt], false, bfrag, (short)0, acc, false, false);
        }
        const int   ncol = nt * 16 + mrow;
        const float bb   = bias[ncol];
        const float vv   = V[ncol];
        // C layout: VGPR r -> row M = khalf*8 + r, col N = mrow.
        #pragma unroll
        for (int r = 0; r < 8; ++r)
            part[r] = fmaf(tanh_hw(acc[r] + bb), vv, part[r]);
    }

    // Reduce the 16 lanes of each half-wave (same rows, 16 different columns each).
    #pragma unroll
    for (int r = 0; r < 8; ++r) {
        float v = part[r];
        v += __shfl_xor(v, 1, 16);
        v += __shfl_xor(v, 2, 16);
        v += __shfl_xor(v, 4, 16);
        v += __shfl_xor(v, 8, 16);
        part[r] = v;
    }
    if (mrow == 0) {
        #pragma unroll
        for (int r = 0; r < 8; ++r)
            logits[row0 + wave * 16 + khalf * 8 + r] = part[r];
    }
}

// ---------------------------------------------------------------------------
// Kernel 3: per-batch softmax statistics (max, sum of exp)
// ---------------------------------------------------------------------------
__global__ __launch_bounds__(256) void stats_kernel(const float* __restrict__ logits,
                                                    float* __restrict__ stats) {
    __shared__ float red[256];
    const int b = blockIdx.x, t = threadIdx.x;
    const float* l = logits + (long)b * SS;

    float m = -3.4e38f;
    for (int s = t; s < SS; s += 256) m = fmaxf(m, l[s]);
    red[t] = m; __syncthreads();
    for (int o = 128; o > 0; o >>= 1) {
        if (t < o) red[t] = fmaxf(red[t], red[t + o]);
        __syncthreads();
    }
    m = red[0]; __syncthreads();

    float sum = 0.f;
    for (int s = t; s < SS; s += 256) sum += __expf(l[s] - m);
    red[t] = sum; __syncthreads();
    for (int o = 128; o > 0; o >>= 1) {
        if (t < o) red[t] += red[t + o];
        __syncthreads();
    }
    if (t == 0) { stats[b * 2] = m; stats[b * 2 + 1] = red[0]; }
}

// ---------------------------------------------------------------------------
// Kernel 4: context[b][d] += sum_s softmax(s) * X[b][s][d]  (split over S)
// ---------------------------------------------------------------------------
__global__ __launch_bounds__(256) void context_kernel(
    const float* __restrict__ X, const float* __restrict__ logits,
    const float* __restrict__ stats, float* __restrict__ out) {
    const int b = blockIdx.x, chunk = blockIdx.y, d = threadIdx.x;
    const float m   = stats[b * 2];
    const float inv = 1.f / stats[b * 2 + 1];

    const long  sbase = (long)chunk * 128;
    const float* l  = logits + (long)b * SS + sbase;
    const float* xp = X + ((long)b * SS + sbase) * DD + d;

    __shared__ float w[128];
    if (d < 128) w[d] = __expf(l[d] - m) * inv;
    __syncthreads();

    float ctx = 0.f;
    for (int i = 0; i < 128; ++i)
        ctx += w[i] * xp[(long)i * DD];

    atomicAdd(out + b * DD + d, ctx);
}

// ---------------------------------------------------------------------------
extern "C" void kernel_launch(void* const* d_in, const int* in_sizes, int n_in,
                              void* d_out, int out_size, void* d_ws, size_t ws_size,
                              hipStream_t stream) {
    const float* X    = (const float*)d_in[0]; // [B,S,D]
    const float* W    = (const float*)d_in[1]; // [D,D]
    const float* bias = (const float*)d_in[2]; // [D]
    const float* V    = (const float*)d_in[3]; // [D,1]
    float* out = (float*)d_out;                // [B,D]

    char* ws = (char*)d_ws;
    _Float16* Wt     = (_Float16*)ws;                         // 128 KB
    float*    logits = (float*)(ws + 131072);                 // 512 KB
    float*    stats  = (float*)(ws + 131072 + 524288);        // 256 B

    hipMemsetAsync(out, 0, (size_t)BB * DD * sizeof(float), stream);

    convw_kernel<<<DD, DD, 0, stream>>>(W, Wt);
    gemm_logits_kernel<<<(BB * SS) / 64, 128, 0, stream>>>(X, Wt, bias, V, logits);
    stats_kernel<<<BB, 256, 0, stream>>>(logits, stats);
    dim3 g(BB, SS / 128);
    context_kernel<<<g, 256, 0, stream>>>(X, logits, stats, out);
}
